// Block_29927332118699
// MI455X (gfx1250) — matrix operand order repaired and checked by reading, not compiled, not run on hardware
//
#include <hip/hip_runtime.h>

typedef __attribute__((ext_vector_type(16))) _Float16 v16h;
typedef __attribute__((ext_vector_type(8)))  float    v8f;
typedef __attribute__((ext_vector_type(4)))  float    f4v;
typedef __attribute__((ext_vector_type(4)))  _Float16 h4v;
typedef __attribute__((ext_vector_type(8)))  _Float16 h8v;

#define HEADS 16
#define DH    64
#define DMODEL 1024
#define SEQ   2048
#define BATCH 2
#define ROWS  (BATCH * SEQ)   // 4096

union HFrag {
    v16h v;
    f4v  f4[2];
};

__device__ __forceinline__ v8f wmma_f16(v16h a, v16h b, v8f c) {
    // D = A(16x32 f16) * B(32x16 f16) + C(16x16 f32)
    return __builtin_amdgcn_wmma_f32_16x16x32_f16(
        /*neg_a=*/false, a, /*neg_b=*/false, b,
        /*c_mod=*/(short)0, c, /*reuse_a=*/false, /*reuse_b=*/false);
}

// Exchange with the opposite 16-lane half (same in-half position) as a pure
// VALU op (v_permlanex16_b32) -- avoids ds_bpermute + s_wait_dscnt.
__device__ __forceinline__ float xor16_f(float x) {
    int i = __float_as_int(x);
    int r = __builtin_amdgcn_permlanex16(i, i, 0x76543210, 0xfedcba98,
                                         /*fi=*/false, /*bound_ctrl=*/false);
    return __int_as_float(r);
}

// ---------------------------------------------------------------------------
// 1) LayerNorm: x (f32 [4096][1024]) -> h (f16 [4096][1024])
// ---------------------------------------------------------------------------
__global__ void ln_kernel(const float* __restrict__ x, const float* __restrict__ g,
                          const float* __restrict__ bta, _Float16* __restrict__ h) {
    int row = blockIdx.x;
    int t = threadIdx.x;               // 256 threads, 4 elems each
    const f4v* xr = (const f4v*)(x + (size_t)row * DMODEL);
    f4v xv = xr[t];
    float s  = xv.x + xv.y + xv.z + xv.w;
    float s2 = xv.x*xv.x + xv.y*xv.y + xv.z*xv.z + xv.w*xv.w;
#pragma unroll
    for (int m = 8; m >= 1; m >>= 1) {
        s  += __shfl_xor(s,  m, 32);
        s2 += __shfl_xor(s2, m, 32);
    }
    s  += xor16_f(s);
    s2 += xor16_f(s2);
    __shared__ float ps[8], ps2[8];
    int lane = t & 31, wv = t >> 5;
    if (lane == 0) { ps[wv] = s; ps2[wv] = s2; }
    __syncthreads();
    float ts = 0.0f, ts2 = 0.0f;
#pragma unroll
    for (int i = 0; i < 8; ++i) { ts += ps[i]; ts2 += ps2[i]; }
    float mu  = ts * (1.0f / DMODEL);
    float var = ts2 * (1.0f / DMODEL) - mu * mu;
    float inv = rsqrtf(var + 1e-5f);
    f4v gv = ((const f4v*)g)[t];
    f4v bv = ((const f4v*)bta)[t];
    h4v o;
    o.x = (_Float16)((xv.x - mu) * inv * gv.x + bv.x);
    o.y = (_Float16)((xv.y - mu) * inv * gv.y + bv.y);
    o.z = (_Float16)((xv.z - mu) * inv * gv.z + bv.z);
    o.w = (_Float16)((xv.w - mu) * inv * gv.w + bv.w);
    ((h4v*)(h + (size_t)row * DMODEL))[t] = o;
}

// ---------------------------------------------------------------------------
// 2) f32 -> f16 weight conversion (vectorized, 4 elems/thread)
// ---------------------------------------------------------------------------
__global__ void cvt_f32_f16(const float* __restrict__ src, _Float16* __restrict__ dst) {
    size_t i = (size_t)blockIdx.x * blockDim.x + threadIdx.x;
    f4v v = ((const f4v*)src)[i];
    h4v o;
    o.x = (_Float16)v.x; o.y = (_Float16)v.y;
    o.z = (_Float16)v.z; o.w = (_Float16)v.w;
    ((h4v*)dst)[i] = o;
}

// ---------------------------------------------------------------------------
// 3/6) WMMA GEMM: D[m][n] = sum_k A[m][k] * W[n][k] (+ bias[n])
//     A: f16 [M][K] row-major, W: f16 [N][K] row-major.
//     One wave computes a 32x32 tile (2x2 WMMA accumulators).
// ---------------------------------------------------------------------------
template <bool OUT_F16>
__global__ void gemm_wmma(const _Float16* __restrict__ A, const _Float16* __restrict__ W,
                          float* __restrict__ Cf, _Float16* __restrict__ Ch,
                          const float* __restrict__ bias, int M, int N, int K) {
    int wave = (int)((blockIdx.x * blockDim.x + threadIdx.x) >> 5);
    int lane = threadIdx.x & 31;
    int nt32 = N >> 5;
    int mg = wave / nt32;
    int ng = wave - mg * nt32;
    if (mg * 32 >= M) return;                  // wave-uniform guard
    int half = lane >> 4;                      // lane group 0/1
    int l15  = lane & 15;

    // A-fragment (16x32): lane m=l15; halves 0-7 -> K = 8*half+0..7,
    //                     halves 8-15 -> K = 16+8*half+0..7
    const _Float16* a0p = A + (size_t)(mg * 32 + l15) * K + half * 8;
    const _Float16* a1p = a0p + (size_t)16 * K;
    // B-fragment (32x16): lane n=l15; halves 0-7 -> K = 8*half+0..7,
    //                     halves 8-15 -> K = 16+8*half+0..7
    const _Float16* b0p = W + (size_t)(ng * 32 + l15) * K + half * 8;
    const _Float16* b1p = b0p + (size_t)16 * K;

    v8f c00 = {}, c01 = {}, c10 = {}, c11 = {};
    for (int k0 = 0; k0 < K; k0 += 32) {
        HFrag a0, a1, b0, b1;
        a0.f4[0] = *(const f4v*)(a0p + k0);
        a0.f4[1] = *(const f4v*)(a0p + k0 + 16);
        a1.f4[0] = *(const f4v*)(a1p + k0);
        a1.f4[1] = *(const f4v*)(a1p + k0 + 16);
        b0.f4[0] = *(const f4v*)(b0p + k0);
        b0.f4[1] = *(const f4v*)(b0p + k0 + 16);
        b1.f4[0] = *(const f4v*)(b1p + k0);
        b1.f4[1] = *(const f4v*)(b1p + k0 + 16);
        c00 = wmma_f16(a0.v, b0.v, c00);
        c01 = wmma_f16(a0.v, b1.v, c01);
        c10 = wmma_f16(a1.v, b0.v, c10);
        c11 = wmma_f16(a1.v, b1.v, c11);
    }
    // C/D layout: lane col n = l15; VGPR v -> row v + 8*half
    int col0 = ng * 32 + l15;
    int col1 = col0 + 16;
    float bz0 = bias ? bias[col0] : 0.0f;
    float bz1 = bias ? bias[col1] : 0.0f;
    int r0 = mg * 32 + half * 8;
#pragma unroll
    for (int v = 0; v < 8; ++v) {
        size_t ra = (size_t)(r0 + v) * N;
        size_t rb = (size_t)(r0 + 16 + v) * N;
        if (OUT_F16) {
            Ch[ra + col0] = (_Float16)(c00[v] + bz0);
            Ch[ra + col1] = (_Float16)(c01[v] + bz1);
            Ch[rb + col0] = (_Float16)(c10[v] + bz0);
            Ch[rb + col1] = (_Float16)(c11[v] + bz1);
        } else {
            Cf[ra + col0] = c00[v] + bz0;
            Cf[ra + col1] = c01[v] + bz1;
            Cf[rb + col0] = c10[v] + bz0;
            Cf[rb + col1] = c11[v] + bz1;
        }
    }
}

// ---------------------------------------------------------------------------
// 4) RoPE + repack. qkv f16 [4096][3072] ->
//    q,k : f16 [B][H][S][64] (q pre-scaled by dh^-0.5)
//    vT  : f16 [B][H][64][S] (transposed so V A-fragments are contiguous)
// One thread per (b,s,h,d<32): handles the (d, d+32) rotation pair.
// ---------------------------------------------------------------------------
__global__ void rope_kernel(const _Float16* __restrict__ qkv,
                            _Float16* __restrict__ q, _Float16* __restrict__ k,
                            _Float16* __restrict__ vT) {
    int idx = blockIdx.x * 256 + threadIdx.x;   // 2^21 threads
    int d = idx & 31;
    int h = (idx >> 5) & 15;
    int s = (idx >> 9) & 2047;
    int b = idx >> 20;
    // inv_freq = 10000^(-d/32) = exp(-d * ln(10000)/32)
    float inv_freq = __expf(-0.28782313662425576f * (float)d);
    float ang = (float)s * inv_freq;
    float sn, cs;
    __sincosf(ang, &sn, &cs);

    size_t row = (size_t)(b * SEQ + s) * (3 * DMODEL);
    int col = h * DH + d;
    float q1 = (float)qkv[row + col];
    float q2 = (float)qkv[row + col + 32];
    float k1 = (float)qkv[row + DMODEL + col];
    float k2 = (float)qkv[row + DMODEL + col + 32];
    float v1 = (float)qkv[row + 2 * DMODEL + col];
    float v2 = (float)qkv[row + 2 * DMODEL + col + 32];

    size_t qdst = ((size_t)(b * HEADS + h) * SEQ + s) * DH + d;
    q[qdst]      = (_Float16)((q1 * cs - q2 * sn) * 0.125f);  // fold dh^-0.5
    q[qdst + 32] = (_Float16)((q2 * cs + q1 * sn) * 0.125f);
    k[qdst]      = (_Float16)(k1 * cs - k2 * sn);
    k[qdst + 32] = (_Float16)(k2 * cs + k1 * sn);

    size_t vdst = ((size_t)(b * HEADS + h) * DH + d) * SEQ + s;
    vT[vdst]            = (_Float16)v1;
    vT[vdst + 32 * SEQ] = (_Float16)v2;
}

// ---------------------------------------------------------------------------
// 5) Flash attention, TRANSPOSED score layout: one wave per 16-query tile.
//    S^T = K * Q^T  => C-layout: key on (VGPR v + half) axis, query on lane.
//    Softmax row stats: per-lane VALU trees + ONE v_permlanex16 exchange.
//    O^T = V^T * P^T => alpha / 1/l are per-lane scalars, packed b128 stores.
// ---------------------------------------------------------------------------
__global__ void attn_kernel(const _Float16* __restrict__ q, const _Float16* __restrict__ k,
                            const _Float16* __restrict__ vT, _Float16* __restrict__ o) {
    __shared__ __align__(16) _Float16 lds[8 * 512];   // 1KB P^T slab per wave
    int lane = threadIdx.x & 31;
    int wv   = threadIdx.x >> 5;
    int gw   = blockIdx.x * 8 + wv;
    if (gw >= BATCH * HEADS * (SEQ / 16)) return;     // wave-uniform
    int qt = gw & 127;            // query tile 0..127
    int hh = (gw >> 7) & 15;      // head
    int bb = gw >> 11;            // batch
    int half = lane >> 4, l15 = lane & 15;

    const _Float16* qh = q  + (size_t)(bb * HEADS + hh) * SEQ * DH;
    const _Float16* kh = k  + (size_t)(bb * HEADS + hh) * SEQ * DH;
    const _Float16* vh = vT + (size_t)(bb * HEADS + hh) * DH * SEQ;
    _Float16* lw = lds + wv * 512;
    int ibase = qt * 16;

    // Q as B-fragments (Q^T): lane n=l15 = query; halves d runs of 8
    HFrag qb0, qb1;
    const _Float16* qp = qh + (size_t)(ibase + l15) * DH + half * 8;
    qb0.f4[0] = *(const f4v*)(qp);        // d = 8h..8h+7
    qb0.f4[1] = *(const f4v*)(qp + 16);   // d = 16+8h..16+8h+7
    qb1.f4[0] = *(const f4v*)(qp + 32);   // d = 32+8h..
    qb1.f4[1] = *(const f4v*)(qp + 48);

    v8f O0 = {}, O1 = {}, O2 = {}, O3 = {};   // O^T: row d (v+8h), col query (l15)
    float mrow = -3.0e38f, lrow = 0.0f;       // per-lane scalars (query l15)

    for (int j = 0; j < SEQ; j += 32) {
        // K as A-fragments: lane m=l15 = key row; halves d runs of 8
        HFrag ka00, ka01, ka10, ka11;   // [keytile][dchunk]
        const _Float16* kp0 = kh + (size_t)(j + l15) * DH + half * 8;
        const _Float16* kp1 = kp0 + 16 * DH;
        ka00.f4[0] = *(const f4v*)(kp0);      ka00.f4[1] = *(const f4v*)(kp0 + 16);
        ka01.f4[0] = *(const f4v*)(kp0 + 32); ka01.f4[1] = *(const f4v*)(kp0 + 48);
        ka10.f4[0] = *(const f4v*)(kp1);      ka10.f4[1] = *(const f4v*)(kp1 + 16);
        ka11.f4[0] = *(const f4v*)(kp1 + 32); ka11.f4[1] = *(const f4v*)(kp1 + 48);

        // S^T tiles: s0 keys j+0..15, s1 keys j+16..31 (rows = keys v+8h)
        v8f s0 = {}, s1 = {};
        s0 = wmma_f16(ka00.v, qb0.v, s0);
        s0 = wmma_f16(ka01.v, qb1.v, s0);
        s1 = wmma_f16(ka10.v, qb0.v, s1);
        s1 = wmma_f16(ka11.v, qb1.v, s1);

        // --- online softmax: per-lane VALU tree + one permlanex16 ---
        float t0 = fmaxf(fmaxf(fmaxf(s0[0], s0[1]), fmaxf(s0[2], s0[3])),
                         fmaxf(fmaxf(s0[4], s0[5]), fmaxf(s0[6], s0[7])));
        float t1 = fmaxf(fmaxf(fmaxf(s1[0], s1[1]), fmaxf(s1[2], s1[3])),
                         fmaxf(fmaxf(s1[4], s1[5]), fmaxf(s1[6], s1[7])));
        float rm = fmaxf(t0, t1);
        rm = fmaxf(rm, xor16_f(rm));                 // combine key halves (VALU)

        float mn = fmaxf(mrow, rm);
        float alpha = __expf(mrow - mn);
        mrow = mn;

        float p0[8], p1[8];
        float rs = 0.0f;
#pragma unroll
        for (int v = 0; v < 8; ++v) {
            p0[v] = __expf(s0[v] - mn);
            p1[v] = __expf(s1[v] - mn);
            rs += p0[v] + p1[v];
        }
        rs += xor16_f(rs);                           // combine key halves (VALU)
        lrow = lrow * alpha + rs;
        O0 *= alpha; O1 *= alpha; O2 *= alpha; O3 *= alpha;

        // --- stage P^T into LDS as [query][32 keys]: packed b128 stores ---
        h8v pk0, pk1;
#pragma unroll
        for (int v = 0; v < 8; ++v) { pk0[v] = (_Float16)p0[v]; pk1[v] = (_Float16)p1[v]; }
        *(h8v*)(lw + l15 * 32 + 8 * half)      = pk0;   // keys 8h+0..7   (tile0)
        *(h8v*)(lw + l15 * 32 + 16 + 8 * half) = pk1;   // keys 16+8h+0..7 (tile1)

        // P^T B-fragment: lane n=l15=query; halves = keys 8h+0..7, 16+8h+0..7
        HFrag pb;
        pb.f4[0] = *(const f4v*)(lw + l15 * 32 + 8 * half);
        pb.f4[1] = *(const f4v*)(lw + l15 * 32 + 16 + 8 * half);

        // V^T A-fragments: lane m=l15 -> d=16t+l15; halves = key runs of 8
        const _Float16* vp = vh + (size_t)l15 * SEQ + j + 8 * half;
        HFrag va0, va1, va2, va3;
        va0.f4[0] = *(const f4v*)(vp);            va0.f4[1] = *(const f4v*)(vp + 16);
        va1.f4[0] = *(const f4v*)(vp + 16 * SEQ); va1.f4[1] = *(const f4v*)(vp + 16 * SEQ + 16);
        va2.f4[0] = *(const f4v*)(vp + 32 * SEQ); va2.f4[1] = *(const f4v*)(vp + 32 * SEQ + 16);
        va3.f4[0] = *(const f4v*)(vp + 48 * SEQ); va3.f4[1] = *(const f4v*)(vp + 48 * SEQ + 16);

        O0 = wmma_f16(va0.v, pb.v, O0);
        O1 = wmma_f16(va1.v, pb.v, O1);
        O2 = wmma_f16(va2.v, pb.v, O2);
        O3 = wmma_f16(va3.v, pb.v, O3);
    }

    // Normalize (per-lane scalar) and store packed 8xf16 per dh-tile.
    float linv = 1.0f / lrow;
    _Float16* op = o + (size_t)(bb * SEQ + ibase + l15) * DMODEL + hh * DH + 8 * half;
    h8v o0, o1, o2, o3;
#pragma unroll
    for (int v = 0; v < 8; ++v) {
        o0[v] = (_Float16)(O0[v] * linv);
        o1[v] = (_Float16)(O1[v] * linv);
        o2[v] = (_Float16)(O2[v] * linv);
        o3[v] = (_Float16)(O3[v] * linv);
    }
    *(h8v*)(op)      = o0;   // d = 8h+0..7
    *(h8v*)(op + 16) = o1;   // d = 16+8h+0..7
    *(h8v*)(op + 32) = o2;
    *(h8v*)(op + 48) = o3;
}

// ---------------------------------------------------------------------------
extern "C" void kernel_launch(void* const* d_in, const int* in_sizes, int n_in,
                              void* d_out, int out_size, void* d_ws, size_t ws_size,
                              hipStream_t stream) {
    (void)in_sizes; (void)n_in; (void)out_size; (void)ws_size;
    const float* x    = (const float*)d_in[0];
    const float* ln_g = (const float*)d_in[1];
    const float* ln_b = (const float*)d_in[2];
    const float* wqkv = (const float*)d_in[3];  // [3072][1024]
    const float* wout = (const float*)d_in[4];  // [1024][1024]
    const float* bout = (const float*)d_in[5];  // [1024]

    char* w = (char*)d_ws;
    size_t off = 0;
    _Float16* hbuf = (_Float16*)(w + off); off += (size_t)ROWS * DMODEL * 2;        // 8 MB
    _Float16* wq   = (_Float16*)(w + off); off += (size_t)3 * DMODEL * DMODEL * 2;  // 6 MB
    _Float16* wo   = (_Float16*)(w + off); off += (size_t)DMODEL * DMODEL * 2;      // 2 MB
    _Float16* qkv  = (_Float16*)(w + off); off += (size_t)ROWS * 3 * DMODEL * 2;    // 24 MB
    _Float16* qb   = (_Float16*)(w + off); off += (size_t)ROWS * DMODEL * 2;        // 8 MB
    _Float16* kb   = (_Float16*)(w + off); off += (size_t)ROWS * DMODEL * 2;        // 8 MB
    _Float16* vT   = (_Float16*)(w + off); off += (size_t)ROWS * DMODEL * 2;        // 8 MB
    _Float16* ob   = (_Float16*)(w + off); off += (size_t)ROWS * DMODEL * 2;        // 8 MB

    // 1) LayerNorm -> f16
    ln_kernel<<<ROWS, 256, 0, stream>>>(x, ln_g, ln_b, hbuf);
    // 2) weights f32 -> f16
    cvt_f32_f16<<<(3 * DMODEL * DMODEL / 4) / 256, 256, 0, stream>>>(wqkv, wq);
    cvt_f32_f16<<<(DMODEL * DMODEL / 4) / 256, 256, 0, stream>>>(wout, wo);
    // 3) QKV GEMM: M=4096, N=3072, K=1024 -> f16 qkv  (12288 waves / 8 per block)
    gemm_wmma<true><<<(ROWS / 32) * (3 * DMODEL / 32) / 8, 256, 0, stream>>>(
        hbuf, wq, nullptr, qkv, nullptr, ROWS, 3 * DMODEL, DMODEL);
    // 4) RoPE + repack (2^21 threads)
    rope_kernel<<<(BATCH * SEQ * HEADS * 32) / 256, 256, 0, stream>>>(qkv, qb, kb, vT);
    // 5) Flash attention (4096 waves, 8 per block)
    attn_kernel<<<(BATCH * HEADS * (SEQ / 16)) / 8, 256, 0, stream>>>(qb, kb, vT, ob);
    // 6) Output GEMM + bias: M=4096, N=1024, K=1024 -> f32 d_out
    gemm_wmma<false><<<(ROWS / 32) * (DMODEL / 32) / 8, 256, 0, stream>>>(
        ob, wo, (float*)d_out, nullptr, bout, ROWS, DMODEL, DMODEL);
}
